// SDGraphEncoder_79310866088277
// MI455X (gfx1250) — compile-verified, matrix operand order checked by reading, not run on hardware
//
#include <hip/hip_runtime.h>
#include <hip/hip_bf16.h>
#include <math.h>

// ---------------------------------------------------------------------------
// SDGraphEncoder forward for MI455X (gfx1250, wave32).
// All MLP/einsum GEMMs (including kNN Gram matrices) run through
// v_wmma_f32_16x16x32_bf16. graph_feature / concat / broadcast operands are
// fused into the GEMM B-loader so the 168MB dense feature tensor is never
// materialized. GEMM is templated on operand mode so the inner loop is
// straight-line; edge tiles use clamped-address + select (no exec branching).
// ---------------------------------------------------------------------------

typedef __attribute__((ext_vector_type(16))) __bf16 v16bf;
typedef __attribute__((ext_vector_type(8)))  float  v8f;

#define INV_BN_C 0.9999950000374997f

__device__ __forceinline__ __bf16 f2bf(float f) { return (__bf16)f; }  // RNE

__device__ __forceinline__ float gelu_t(float x) {
  // jax.nn.gelu default (approximate=True, tanh form)
  float x3 = x * x * x;
  return 0.5f * x * (1.0f + tanhf(0.7978845608028654f * (x + 0.044715f * x3)));
}

// ---------------------------------------------------------------------------
// Batched WMMA GEMM:  Y[b] = act( A * B(X[b]) + bias )
//   A: M x K row-major weights (TRANSA: A(m,k) = A[k*lda + m])
//   BMODE 0: plain            B(k,col) = B0[k*ldb + col]
//   BMODE 1: graph feature    col=(n,j): k<C1 ? B0[k][idx[n,j]]-B0[k][n]
//                                             : B0[k-C1][n]
//   BMODE 2: channel concat   k<C1 ? B0[k][col] : B1[k-C1][col]   (same ldb)
//   BMODE 3: concat+broadcast k<C1 ? B0[k][col] : B1[k-C1][col/bdiv]
//   ACT: BN-fold (*INV_BN*g + bt) then GELU (bias applied only when ACT)
// Block: 8 waves, each wave owns a 16x16 C tile (block tile 32x64), K-step 32.
// Fragment layouts per CDNA5 ISA 7.12.2. C1 is always a multiple of 32 here,
// so per-K-step side selection is wave-uniform in the fast path.
// ---------------------------------------------------------------------------
template<int BMODE, bool TRANSA, bool ACT>
__global__ __launch_bounds__(256) void wmma_gemm_t(
    const float* __restrict__ A, int lda, long long strideA,
    const float* __restrict__ B0, long long strideB0,
    const float* __restrict__ B1, long long strideB1,
    int ldb, int C1, const int* __restrict__ idxp, long long strideIdx,
    int kk, int bdiv,
    const float* __restrict__ bias, const float* __restrict__ g,
    const float* __restrict__ bt,
    float* __restrict__ Y, long long strideY, int M, int N, int K)
{
  const int b = blockIdx.z;
  A  += (long long)b * strideA;
  B0 += (long long)b * strideB0;
  if (BMODE == 2 || BMODE == 3) B1   += (long long)b * strideB1;
  if (BMODE == 1)               idxp += (long long)b * strideIdx;
  Y  += (long long)b * strideY;

  const int lane  = threadIdx.x;       // 0..31
  const int wave  = threadIdx.y;       // 0..7
  const int mBase = blockIdx.x * 32 + (wave >> 2) * 16;
  const int nBase = blockIdx.y * 64 + (wave & 3) * 16;
  const int lrow  = lane & 15;
  const int hi    = lane >> 4;

  const int mRow = mBase + lrow;       // A row supplied by this lane
  const int nCol = nBase + lrow;       // B col supplied by this lane
  const int mRc  = mRow < M ? mRow : M - 1;   // clamped (edge path)
  const int nCc  = nCol < N ? nCol : N - 1;

  int gn = 0, nbr = 0, bc = 0;
  if (BMODE == 1) {                    // graph-feature column decomposition
    gn  = nCc / kk;
    nbr = idxp[gn * kk + (nCc - gn * kk)];
  }
  if (BMODE == 3) bc = nCc / bdiv;
  const int ldb1 = (BMODE == 3) ? (N / bdiv) : 0;

  v8f acc = {};
  int kb = 0;

  // ---- wave-uniform fast path: fully interior tile, full K steps ----------
  if (mBase + 16 <= M && nBase + 16 <= N) {
    const int kFull = K & ~31;
    const float* aP = TRANSA ? (A + mRow)
                             : (A + (long long)mRow * lda + (hi << 3));
    for (; kb < kFull; kb += 32) {
      if (BMODE == 0 && kb + 32 < K)
        __builtin_prefetch(B0 + (long long)(kb + 32) * ldb + nCol, 0, 1);

      v16bf afr, bfr;
#pragma unroll
      for (int h = 0; h < 16; ++h) {
        const int ko = (h < 8 ? h : h + 8);
        float av = TRANSA ? aP[(long long)(kb + ko + (hi << 3)) * lda]
                          : aP[kb + ko];
        afr[h] = f2bf(av);
      }

      if (BMODE == 0) {
#pragma unroll
        for (int h = 0; h < 16; ++h) {
          const int k = kb + (h < 8 ? h : h + 8) + (hi << 3);
          bfr[h] = f2bf(B0[(long long)k * ldb + nCol]);
        }
      } else if (BMODE == 1) {
        if (kb < C1) {
#pragma unroll
          for (int h = 0; h < 16; ++h) {
            const int k = kb + (h < 8 ? h : h + 8) + (hi << 3);
            bfr[h] = f2bf(B0[(long long)k * ldb + nbr] -
                          B0[(long long)k * ldb + gn]);
          }
        } else {
#pragma unroll
          for (int h = 0; h < 16; ++h) {
            const int k = kb - C1 + (h < 8 ? h : h + 8) + (hi << 3);
            bfr[h] = f2bf(B0[(long long)k * ldb + gn]);
          }
        }
      } else if (BMODE == 2) {
        const float* Bs = (kb < C1) ? B0 : B1;
        const int ks = (kb < C1) ? kb : kb - C1;
#pragma unroll
        for (int h = 0; h < 16; ++h) {
          const int k = ks + (h < 8 ? h : h + 8) + (hi << 3);
          bfr[h] = f2bf(Bs[(long long)k * ldb + nCol]);
        }
      } else {  // BMODE == 3
        if (kb < C1) {
#pragma unroll
          for (int h = 0; h < 16; ++h) {
            const int k = kb + (h < 8 ? h : h + 8) + (hi << 3);
            bfr[h] = f2bf(B0[(long long)k * ldb + nCol]);
          }
        } else {
#pragma unroll
          for (int h = 0; h < 16; ++h) {
            const int k = kb - C1 + (h < 8 ? h : h + 8) + (hi << 3);
            bfr[h] = f2bf(B1[(long long)k * ldb1 + bc]);
          }
        }
      }
      acc = __builtin_amdgcn_wmma_f32_16x16x32_bf16(false, afr, false, bfr,
                                                    (short)0, acc, false, false);
    }
  }

  // ---- edge/tail path: clamped addresses + selects (branch-free) ----------
  for (; kb < K; kb += 32) {
    v16bf afr, bfr;
#pragma unroll
    for (int h = 0; h < 16; ++h) {
      const int k  = kb + (h < 8 ? h : h + 8) + (hi << 3);
      const int kc = k < K ? k : K - 1;
      float av = TRANSA ? A[(long long)kc * lda + mRc]
                        : A[(long long)mRc * lda + kc];
      av = (k < K && mRow < M) ? av : 0.0f;
      afr[h] = f2bf(av);

      float bv;
      if (BMODE == 0) {
        bv = B0[(long long)kc * ldb + nCc];
      } else if (BMODE == 1) {
        bv = (kc < C1) ? (B0[(long long)kc * ldb + nbr] -
                          B0[(long long)kc * ldb + gn])
                       : B0[(long long)(kc - C1) * ldb + gn];
      } else if (BMODE == 2) {
        bv = (kc < C1) ? B0[(long long)kc * ldb + nCc]
                       : B1[(long long)(kc - C1) * ldb + nCc];
      } else {
        bv = (kc < C1) ? B0[(long long)kc * ldb + nCc]
                       : B1[(long long)(kc - C1) * ldb1 + bc];
      }
      bv = (k < K && nCol < N) ? bv : 0.0f;
      bfr[h] = f2bf(bv);
    }
    acc = __builtin_amdgcn_wmma_f32_16x16x32_bf16(false, afr, false, bfr,
                                                  (short)0, acc, false, false);
  }

  if (nCol >= N) return;
#pragma unroll
  for (int r = 0; r < 8; ++r) {
    const int row = mBase + r + (hi << 3);   // ISA C/D layout
    if (row >= M) continue;
    float v = acc[r];
    if (ACT) {
      v += bias[row];
      v = gelu_t(v * INV_BN_C * g[row] + bt[row]);
    }
    Y[(long long)row * N + nCol] = v;
  }
}

// ---------------------------------------------------------------------------
// kNN top-k from Gram matrix:  dist(n,m) = G[n,n] - 2 G[n,m] + G[m,m]
// Tie-break matches jax.lax.top_k (lower index first): strict '<', ascending m.
// ---------------------------------------------------------------------------
__global__ void knn_topk(const float* __restrict__ gram, int npts, int kk,
                         int* __restrict__ out)
{
  const int b = blockIdx.y;
  const int n = blockIdx.x * blockDim.x + threadIdx.x;
  if (n >= npts) return;
  const float* G = gram + (long long)b * npts * npts;
  const float gnn = G[(long long)n * npts + n];

  float bd[10]; int bi[10];
  for (int j = 0; j < kk; ++j) { bd[j] = 3.4e38f; bi[j] = 0; }
  for (int m = 0; m < npts; ++m) {
    float d = gnn - 2.0f * G[(long long)n * npts + m] + G[(long long)m * npts + m];
    if (d < bd[kk - 1]) {
      int p = kk - 1;
      while (p > 0 && d < bd[p - 1]) { bd[p] = bd[p - 1]; bi[p] = bi[p - 1]; --p; }
      bd[p] = d; bi[p] = m;
    }
  }
  int* o = out + ((long long)b * npts + n) * kk;
  for (int j = 0; j < kk; ++j) o[j] = bi[j];
}

// max over trailing axis of size kk
__global__ void maxlast(const float* __restrict__ in, float* __restrict__ out,
                        long long outer, int kk)
{
  long long i = blockIdx.x * (long long)blockDim.x + threadIdx.x;
  if (i >= outer) return;
  const float* p = in + i * kk;
  float m = p[0];
  for (int j = 1; j < kk; ++j) m = fmaxf(m, p[j]);
  out[i] = m;
}

// circular-pad conv1d + BN-fold + GELU (direct form; used for d2s_temp & down)
__global__ void conv1d_cpad_bn_gelu(
    const float* __restrict__ x, const float* __restrict__ w,
    const float* __restrict__ bias, const float* __restrict__ g,
    const float* __restrict__ bt, float* __restrict__ y,
    int B, int Cin, int Cout, int L, int ksz, int stride, int pad, int Lout)
{
  long long tid = blockIdx.x * (long long)blockDim.x + threadIdx.x;
  long long total = (long long)B * Cout * Lout;
  if (tid >= total) return;
  int t = (int)(tid % Lout);
  long long r = tid / Lout;
  int o = (int)(r % Cout);
  int b = (int)(r / Cout);

  const float* xb = x + (long long)b * Cin * L;
  const float* wo = w + (long long)o * Cin * ksz;
  float acc = bias[o];
  for (int i = 0; i < Cin; ++i) {
    const float* xi = xb + (long long)i * L;
    const float* wi = wo + i * ksz;
    for (int q = 0; q < ksz; ++q) {
      int pos = t * stride + q - pad;
      if (pos < 0) pos += L;
      if (pos >= L) pos -= L;
      acc += wi[q] * xi[pos];
    }
  }
  acc = acc * INV_BN_C * g[o] + bt[o];
  y[tid] = gelu_t(acc);
}

// ---------------------------------------------------------------------------
// Host-side helpers
// ---------------------------------------------------------------------------
static inline void launch_gemm(hipStream_t s,
    const float* A, int lda, int transA, long long sA,
    const float* B0, long long sB0, const float* B1, long long sB1,
    int ldb, int C1, int bmode, const int* idx, long long sIdx, int kk, int bdiv,
    const float* bias, const float* g, const float* bt, int act,
    float* Y, long long sY, int M, int N, int K, int batch)
{
  (void)act;
  dim3 grid((unsigned)((M + 31) / 32), (unsigned)((N + 63) / 64), (unsigned)batch);
  dim3 blk(32, 8, 1);
#define LG(BM, TR, AC)                                                       \
  wmma_gemm_t<BM, TR, AC><<<grid, blk, 0, s>>>(A, lda, sA, B0, sB0, B1, sB1, \
      ldb, C1, idx, sIdx, kk, bdiv, bias, g, bt, Y, sY, M, N, K)
  if (transA)           LG(0, true,  false);   // Gram (no bias, no act)
  else if (bmode == 0)  LG(0, false, true);
  else if (bmode == 1)  LG(1, false, true);
  else if (bmode == 2)  LG(2, false, true);
  else                  LG(3, false, true);
#undef LG
}

// One GCN encoder. Param leaves (JAX pytree, dicts key-sorted => b,bt,g,w):
// pb+0..3 conv1.l0, +4..7 conv1.l1, +8..11 conv2.l0, +12..15 conv2.l1,
// +16..19 conv3.l0, +20..23 conv3.l1.  Channels: 512->362->256 per stack.
static void run_gcn(hipStream_t s, const float* const* prm, int pb,
                    const float* x, int npts, int kk,
                    float* gram, int* idx, float* h1, float* h2,
                    float* x1, float* x2, float* h3,
                    float* out, long long sOut)
{
  const int B = 8, C = 256;
  const long long sX = (long long)C * npts;
  const int N1 = npts * kk;

  // kNN #1: Gram = X^T X via WMMA (transA), then top-k
  launch_gemm(s, x, npts, 1, sX, x, sX, nullptr, 0, npts, 0, 0, nullptr, 0, 0, 1,
              nullptr, nullptr, nullptr, 0, gram, (long long)npts * npts,
              npts, npts, C, B);
  { dim3 g2((npts + 127) / 128, B); knn_topk<<<g2, 128, 0, s>>>(gram, npts, kk, idx); }

  // conv1 (graph-feature fused into B operand)
  launch_gemm(s, prm[pb + 3], 512, 0, 0, x, sX, nullptr, 0, npts, C, 1, idx,
              (long long)npts * kk, kk, 1,
              prm[pb + 0], prm[pb + 2], prm[pb + 1], 1, h1, 362LL * N1, 362, N1, 512, B);
  launch_gemm(s, prm[pb + 7], 362, 0, 0, h1, 362LL * N1, nullptr, 0, N1, 0, 0,
              nullptr, 0, 0, 1,
              prm[pb + 4], prm[pb + 6], prm[pb + 5], 1, h2, 256LL * N1, 256, N1, 362, B);
  { long long outer = (long long)B * 256 * npts;
    maxlast<<<dim3((unsigned)((outer + 255) / 256)), 256, 0, s>>>(h2, x1, outer, kk); }

  // kNN #2 on x1
  launch_gemm(s, x1, npts, 1, sX, x1, sX, nullptr, 0, npts, 0, 0, nullptr, 0, 0, 1,
              nullptr, nullptr, nullptr, 0, gram, (long long)npts * npts,
              npts, npts, C, B);
  { dim3 g2((npts + 127) / 128, B); knn_topk<<<g2, 128, 0, s>>>(gram, npts, kk, idx); }

  // conv2
  launch_gemm(s, prm[pb + 11], 512, 0, 0, x1, sX, nullptr, 0, npts, C, 1, idx,
              (long long)npts * kk, kk, 1,
              prm[pb + 8], prm[pb + 10], prm[pb + 9], 1, h1, 362LL * N1, 362, N1, 512, B);
  launch_gemm(s, prm[pb + 15], 362, 0, 0, h1, 362LL * N1, nullptr, 0, N1, 0, 0,
              nullptr, 0, 0, 1,
              prm[pb + 12], prm[pb + 14], prm[pb + 13], 1, h2, 256LL * N1, 256, N1, 362, B);
  { long long outer = (long long)B * 256 * npts;
    maxlast<<<dim3((unsigned)((outer + 255) / 256)), 256, 0, s>>>(h2, x2, outer, kk); }

  // conv3 on concat(x1, x2) (fused)
  launch_gemm(s, prm[pb + 19], 512, 0, 0, x1, sX, x2, sX, npts, C, 2,
              nullptr, 0, 0, 1,
              prm[pb + 16], prm[pb + 18], prm[pb + 17], 1, h3, 362LL * npts,
              362, npts, 512, B);
  launch_gemm(s, prm[pb + 23], 362, 0, 0, h3, 362LL * npts, nullptr, 0, npts, 0, 0,
              nullptr, 0, 0, 1,
              prm[pb + 20], prm[pb + 22], prm[pb + 21], 1, out, sOut, 256, npts, 362, B);
}

extern "C" void kernel_launch(void* const* d_in, const int* in_sizes, int n_in,
                              void* d_out, int out_size, void* d_ws, size_t ws_size,
                              hipStream_t stream)
{
  (void)in_sizes; (void)n_in; (void)out_size; (void)ws_size;

  const float* const* prm = (const float* const*)d_in;
  const float* sparse = prm[0];   // (8,128,32)
  const float* dense  = prm[1];   // (8,128,32,32)
  // Param leaf order (JAX pytree: dict keys sorted, lists in order, leaf dicts
  // {b,bt,g,w}): d2s_enc[2..5], d2s_temp[6..9], dn_gcn[10..33], down[34..37],
  // s2d_enc[38..41], sp_gcn[42..65].

  // Workspace arena (~288 MB of f32; assumes ws_size is sufficient)
  float* ws = (float*)d_ws;
  size_t off = 0;
  auto alloc = [&](size_t n) { float* p = ws + off; off += n; return p; };
  float* convt   = alloc(8ull * 128 * 1024);
  float* spfd    = alloc(8ull * 128 * 32);
  float* usp0    = alloc(8ull * 256 * 32);
  float* udn0    = alloc(8ull * 256 * 1024);
  float* gram_sp = alloc(8ull * 32 * 32);
  int*   idx_sp  = (int*)alloc(8ull * 32 * 2);
  float* hsp1    = alloc(8ull * 362 * 64);
  float* hsp2    = alloc(8ull * 256 * 64);
  float* x1s     = alloc(8ull * 256 * 32);
  float* x2s     = alloc(8ull * 256 * 32);
  float* hsp3    = alloc(8ull * 362 * 32);
  float* gram_dn = alloc(8ull * 1024 * 1024);
  int*   idx_dn  = (int*)alloc(8ull * 1024 * 10);
  float* hdn1    = alloc(8ull * 362 * 10240);
  float* hdn2    = alloc(8ull * 256 * 10240);
  float* x1d     = alloc(8ull * 256 * 1024);
  float* x2d     = alloc(8ull * 256 * 1024);
  float* hdn3    = alloc(8ull * 362 * 1024);
  float* udn     = alloc(8ull * 256 * 1024);

  // 1) d2s_temp: circular-pad conv1d(k=5,s=1) + BN + GELU on dense (8,128,1024)
  { long long total = 8LL * 128 * 1024;
    conv1d_cpad_bn_gelu<<<dim3((unsigned)((total + 255) / 256)), 256, 0, stream>>>(
        dense, prm[9], prm[6], prm[8], prm[7], convt,
        8, 128, 128, 1024, 5, 1, 2, 1024); }

  // 2) sp_from_dn: max over last 32 points
  { long long outer = 8LL * 128 * 32;
    maxlast<<<dim3((unsigned)((outer + 255) / 256)), 256, 0, stream>>>(
        convt, spfd, outer, 32); }

  // 3) d2s_enc: MLP(256->256) on concat(sparse, sp_from_dn)  -> u_sp0
  launch_gemm(stream, prm[5], 256, 0, 0, sparse, 128LL * 32, spfd, 128LL * 32,
              32, 128, 2, nullptr, 0, 0, 1,
              prm[2], prm[4], prm[3], 1, usp0, 256LL * 32, 256, 32, 256, 8);

  // 4) s2d_enc: MLP(256->256) on concat(dense, broadcast(sparse)) -> u_dn0
  launch_gemm(stream, prm[41], 256, 0, 0, dense, 128LL * 1024, sparse, 128LL * 32,
              1024, 128, 3, nullptr, 0, 0, 32,
              prm[38], prm[40], prm[39], 1, udn0, 256LL * 1024, 256, 1024, 256, 8);

  // 5) sparse GCN (npts=32, k=2) -> u_sp output (first 65536 floats of d_out)
  run_gcn(stream, prm, 42, usp0, 32, 2, gram_sp, idx_sp, hsp1, hsp2,
          x1s, x2s, hsp3, (float*)d_out, 256LL * 32);

  // 6) dense GCN (npts=1024, k=10) -> u_dn
  run_gcn(stream, prm, 10, udn0, 1024, 10, gram_dn, idx_dn, hdn1, hdn2,
          x1d, x2d, hdn3, udn, 256LL * 1024);

  // 7) down: circular-pad conv1d(k=6,s=2) + BN + GELU -> d_out tail (8,256,512)
  { long long total = 8LL * 256 * 512;
    conv1d_cpad_bn_gelu<<<dim3((unsigned)((total + 255) / 256)), 256, 0, stream>>>(
        udn, prm[37], prm[34], prm[36], prm[35],
        ((float*)d_out) + 8 * 256 * 32,
        8, 256, 256, 1024, 6, 2, 2, 512); }
}
